// SmilesDecoder_7026566496689
// MI455X (gfx1250) — matrix operand order repaired and checked by reading, not compiled, not run on hardware
//
#include <hip/hip_runtime.h>
#include <hip/hip_bf16.h>

// ---------------------------------------------------------------------------
// Problem constants
// ---------------------------------------------------------------------------
#define Hc   501            // hidden size (real)
#define Tc   200            // timesteps
#define Vc   71             // vocab
#define Bc   256            // batch
#define HP   512            // hidden padded to WMMA K/N multiple
#define GP   1536           // 3 * HP  (padded gate columns)
#define VP   80             // vocab padded to 16
#define NROW (Bc * Tc)      // 51200 flattened rows for big GEMMs

typedef __bf16 bf16;
typedef unsigned int u32;
typedef unsigned long long u64;
typedef __attribute__((ext_vector_type(16))) __bf16 v16bf;
typedef __attribute__((ext_vector_type(8)))  __bf16 v8bf;
typedef __attribute__((ext_vector_type(8)))  float  v8f;
typedef __attribute__((ext_vector_type(4)))  u32    v4u;
typedef __attribute__((ext_vector_type(4)))  int    v4i;
typedef __attribute__((ext_vector_type(8)))  int    v8i;

#define DEVINL __device__ __forceinline__

// ---------------------------------------------------------------------------
// Feature detection: Tensor Data Mover builtin (probe-verified on both
// toolchains; arity differs -> discriminate via the therock-only TDM header)
// ---------------------------------------------------------------------------
#if defined(__has_builtin)
#  if __has_builtin(__builtin_amdgcn_tensor_load_to_lds)
#    define HAVE_TDM 1
#  endif
#endif
#ifndef HAVE_TDM
#  define HAVE_TDM 0
#endif

// ---------------------------------------------------------------------------
// WMMA helpers (CDNA5 gfx1250, wave32)
// D = A(16x32 bf16) * B(32x16 bf16) + C(16x16 f32)
// ---------------------------------------------------------------------------
DEVINL v8f wmma_bf16(v16bf a, v16bf b, v8f c) {
  return __builtin_amdgcn_wmma_f32_16x16x32_bf16(
      /*neg_a=*/false, a, /*neg_b=*/false, b,
      /*c_mod=*/(short)0, c, /*reuse_a=*/false, /*reuse_b=*/false);
}

// A-fragment, ISA 16-bit A 16x32 layout:
// lanes 0-15  : row = lane,    K = kb+{0..7}  and kb+{16..23}
// lanes 16-31 : row = lane-16, K = kb+{8..15} and kb+{24..31}
DEVINL v16bf load_a_frag(const bf16* base, int ldk, int kb, int lane) {
  const int half = lane >> 4;
  const int row  = lane & 15;
  const bf16* p = base + row * ldk + kb + half * 8;
  v8bf lo = *(const v8bf*)(p);
  v8bf hi = *(const v8bf*)(p + 16);
  v16bf r;
#pragma unroll
  for (int i = 0; i < 8; ++i) { r[i] = lo[i]; r[i + 8] = hi[i]; }
  return r;
}

// B-fragment, ISA 16-bit B 32x16 layout. W is row-major [N, 512] so W-row n
// supplies B-column n (B[k][n] = W[n][k]):
// lanes 0-15  : col = lane,    K = kb+{0..7}  and kb+{16..23}
// lanes 16-31 : col = lane-16, K = kb+{8..15} and kb+{24..31}
DEVINL v16bf load_b_frag(const bf16* __restrict__ W, int nbase, int kb, int lane) {
  const int col  = lane & 15;
  const int koff = (lane >> 4) * 8;
  const bf16* p = W + (size_t)(nbase + col) * HP + kb + koff;
  v8bf lo = *(const v8bf*)(p);
  v8bf hi = *(const v8bf*)(p + 16);
  v16bf r;
#pragma unroll
  for (int i = 0; i < 8; ++i) { r[i] = lo[i]; r[i + 8] = hi[i]; }
  return r;
}

DEVINL float sigmoidf_(float x) { return 1.f / (1.f + __expf(-x)); }

// ---------------------------------------------------------------------------
// Tensor Data Mover: build D# (descriptor) per CDNA5 ISA
// §8.3/8.4 and issue TENSOR_LOAD_TO_LDS.  2D tile: tile_d0 elements per row,
// tile_d1 rows, row stride = stride0 elements, element size = 1<<dsz_log B.
// ---------------------------------------------------------------------------
DEVINL void tdm_load_tile(u32 lds_off, const void* gsrc,
                          u32 tensor_d0, u32 tensor_d1,
                          u32 tile_d0, u32 tile_d1,
                          u64 stride0, u32 dsz_log) {
#if HAVE_TDM
  u64 ga = (u64)(uintptr_t)gsrc;
  v4u g0;
  g0[0] = 1u;                                   // count=1, user mode
  g0[1] = lds_off;                              // lds_addr (bytes)
  g0[2] = (u32)ga;                              // global_addr[31:0]
  g0[3] = (u32)((ga >> 32) & 0x01FFFFFFull)     // global_addr[56:32]
          | (2u << 30);                         // type = 2 ("image")
  v8i g1;
  g1[0] = (int)(dsz_log << 16);                 // wg_mask=0, data_size
  g1[1] = (int)((tensor_d0 & 0xFFFFu) << 16);   // tensor_dim0[15:0] @bit48
  g1[2] = (int)(((tensor_d0 >> 16) & 0xFFFFu)   // tensor_dim0[31:16]
          | ((tensor_d1 & 0xFFFFu) << 16));     // tensor_dim1[15:0]
  g1[3] = (int)(((tensor_d1 >> 16) & 0xFFFFu)   // tensor_dim1[31:16]
          | ((tile_d0 & 0xFFFFu) << 16));       // tile_dim0
  g1[4] = (int)(tile_d1 & 0xFFFFu);             // tile_dim1 (tile_dim2=0)
  g1[5] = (int)(u32)(stride0 & 0xFFFFFFFFull);  // tensor_dim0_stride[31:0]
  g1[6] = (int)(u32)((stride0 >> 32) & 0xFFFFull); // stride[47:32], s1 lo = 0
  g1[7] = 0;                                    // tensor_dim1_stride hi
  v4i z4 = {0, 0, 0, 0};
#if __has_include(<hip/amd_detail/amd_gfx1250_TDM.h>)
  v8i z8 = {0, 0, 0, 0, 0, 0, 0, 0};
  __builtin_amdgcn_tensor_load_to_lds(g0, g1, z4, z4, z8, 0);   // clang-23 form
#else
  __builtin_amdgcn_tensor_load_to_lds(g0, g1, z4, z4, 0);       // ROCm 7.2 form
#endif
#else
  (void)lds_off; (void)gsrc; (void)tensor_d0; (void)tensor_d1;
  (void)tile_d0; (void)tile_d1; (void)stride0; (void)dsz_log;
#endif
}

DEVINL void tdm_wait0() {
#if defined(__has_builtin) && __has_builtin(__builtin_amdgcn_s_wait_tensorcnt)
  __builtin_amdgcn_s_wait_tensorcnt(0);
#else
  asm volatile("s_wait_tensorcnt 0x0" ::: "memory");
#endif
}

// TDM ops of one wave complete IN ORDER: waiting tensorcnt<=1 while the most
// recently issued load (next step's x tile) is still in flight guarantees all
// earlier loads (this step's h tile) have landed in LDS.
DEVINL void tdm_wait1() {
#if defined(__has_builtin) && __has_builtin(__builtin_amdgcn_s_wait_tensorcnt)
  __builtin_amdgcn_s_wait_tensorcnt(1);
#else
  asm volatile("s_wait_tensorcnt 0x1" ::: "memory");
#endif
}

// ---------------------------------------------------------------------------
// Weight / bias repack kernels (f32 -> zero-padded bf16 in WMMA-friendly
// [gate*512 + j, 512] row-major layout)
// ---------------------------------------------------------------------------
__global__ void pack3_kernel(const float* __restrict__ src,  // [3*Hc, Kreal]
                             bf16* __restrict__ dst,         // [GP, HP]
                             int Kreal) {
  int idx = blockIdx.x * blockDim.x + threadIdx.x;
  if (idx >= GP * HP) return;
  int n = idx >> 9, k = idx & 511;
  int g = n >> 9, j = n & 511;
  float v = 0.f;
  if (j < Hc && k < Kreal) v = src[(size_t)(g * Hc + j) * Kreal + k];
  dst[idx] = (bf16)v;
}

__global__ void packwd_kernel(const float* __restrict__ src,  // [Vc, Hc]
                              bf16* __restrict__ dst) {       // [VP, HP]
  int idx = blockIdx.x * blockDim.x + threadIdx.x;
  if (idx >= VP * HP) return;
  int n = idx >> 9, k = idx & 511;
  float v = 0.f;
  if (n < Vc && k < Hc) v = src[(size_t)n * Hc + k];
  dst[idx] = (bf16)v;
}

__global__ void packbias_kernel(const float* __restrict__ src,  // [3*Hc]
                                float* __restrict__ dst) {      // [GP]
  int idx = blockIdx.x * blockDim.x + threadIdx.x;
  if (idx >= GP) return;
  int g = idx >> 9, j = idx & 511;
  dst[idx] = (j < Hc) ? src[g * Hc + j] : 0.f;
}

// Layer-1 input projection is time invariant: gi1[b, n] = x[b]*W_ih1[n] + b_ih1[n]
__global__ void gi1_kernel(const float* __restrict__ x,      // [B,1]
                           const float* __restrict__ Wih1,   // [3*Hc, 1]
                           const float* __restrict__ bih1,   // [3*Hc]
                           float* __restrict__ gi1) {        // [B, GP]
  int idx = blockIdx.x * blockDim.x + threadIdx.x;
  if (idx >= Bc * GP) return;
  int b = idx / GP, n = idx % GP;
  int g = n >> 9, j = n & 511;
  gi1[idx] = (j < Hc) ? (x[b] * Wih1[g * Hc + j] + bih1[g * Hc + j]) : 0.f;
}

__global__ void zeroh_kernel(bf16* __restrict__ Hbuf) {  // [2, B, HP]
  int idx = blockIdx.x * blockDim.x + threadIdx.x;
  if (idx < 2 * Bc * HP) Hbuf[idx] = (bf16)0.f;
}

// ---------------------------------------------------------------------------
// GRU recurrence. Grid = (B/16 batch tiles, 8 N-slices), 128 threads (4 waves).
// Wave w of slice s owns padded-h columns j0 = s*64 + w*16 .. +15 and computes
// gate tiles at N-offsets {j0, 512+j0, 1024+j0}. h is exchanged through a
// ping-pong global buffer; per-step sync uses the CDNA5 cluster split-barrier.
// TDM pipeline: x_{t+1} DMA (into the opposite Xsh buffer) stays in flight
// across the whole compute phase; s_wait_tensorcnt 1 + in-order TDM completion
// guarantees h_t / x_t are resident.
// ---------------------------------------------------------------------------
template <bool HAS_X>
__global__ __launch_bounds__(128) void gru_seq_kernel(
    const float* __restrict__ gi1,   // [B, GP]   (layer 1 only)
    const bf16* __restrict__ Xseq,   // [B, T, HP] (layers 2/3; tanh of prev layer)
    const bf16* __restrict__ WihP,   // [GP, HP]
    const bf16* __restrict__ WhhP,   // [GP, HP]
    const float* __restrict__ bihP,  // [GP]
    const float* __restrict__ bhhP,  // [GP]
    bf16* __restrict__ Hbuf,         // [2, B, HP] ping-pong carried hidden
    bf16* __restrict__ Oseq)         // [B, T, HP] tanh(h_t) layer output
{
  __shared__ bf16 Hsh[16 * HP];
  __shared__ bf16 Xsh[2][16 * HP];

  const int lane = threadIdx.x & 31;
  const int wave = threadIdx.x >> 5;
  const int m0   = blockIdx.x * 16;              // batch tile
  const int j0   = blockIdx.y * 64 + wave * 16;  // padded-h column tile
  const int half = lane >> 4;
  const int jc   = j0 + (lane & 15);             // this lane's N column

#if HAVE_TDM
  // Prologue: start the x_0 DMA before entering the scan.
  if (HAS_X && threadIdx.x < 32) {
    tdm_load_tile((u32)(uintptr_t)&Xsh[0][0],
                  Xseq + (size_t)m0 * Tc * HP,
                  /*tensor_d0=*/HP, /*tensor_d1=*/0x40000000u,
                  /*tile_d0=*/HP, /*tile_d1=*/16,
                  /*stride0=*/(u64)Tc * HP, /*dsz_log=*/1);
  }
#endif

  for (int t = 0; t < Tc; ++t) {
    const bf16* Hsrc = Hbuf + (size_t)(t & 1) * (Bc * HP) + (size_t)m0 * HP;
    const bf16* Xcur = &Xsh[t & 1][0];
#if HAVE_TDM
    if (threadIdx.x < 32) {
      // h_t tile: contiguous 16 KB -> one row of 2048 x 8B elements.
      // Safe to overwrite Hsh: every wave passed last step's __syncthreads.
      tdm_load_tile((u32)(uintptr_t)Hsh, Hsrc,
                    /*tensor_d0=*/2048, /*tensor_d1=*/1,
                    /*tile_d0=*/2048, /*tile_d1=*/0,
                    /*stride0=*/2048, /*dsz_log=*/3);
      if (HAS_X && t + 1 < Tc) {
        // Pipeline: kick off x_{t+1} into the opposite buffer, then only
        // require the OLDER loads (h_t, x_t) to have completed.
        const bf16* Xnext = Xseq + ((size_t)m0 * Tc + (t + 1)) * HP;
        tdm_load_tile((u32)(uintptr_t)&Xsh[(t + 1) & 1][0], Xnext,
                      /*tensor_d0=*/HP, /*tensor_d1=*/0x40000000u,
                      /*tile_d0=*/HP, /*tile_d1=*/16,
                      /*stride0=*/(u64)Tc * HP, /*dsz_log=*/1);
        tdm_wait1();   // x_{t+1} still in flight during the WMMA loop
      } else {
        tdm_wait0();
      }
    }
#else
    for (int i = threadIdx.x; i < 1024; i += 128)
      *((v8bf*)Hsh + i) = *((const v8bf*)Hsrc + i);
    if (HAS_X) {
      for (int i = threadIdx.x; i < 1024; i += 128) {
        int r = i >> 6, c = i & 63;
        const v8bf* src =
            (const v8bf*)(Xseq + ((size_t)(m0 + r) * Tc + t) * HP) + c;
        *((v8bf*)(bf16*)Xcur + i) = *src;
      }
    }
    if (HAS_X && t + 1 < Tc)  // warm L2 for the next timestep's x tile
      __builtin_prefetch(Xseq + ((size_t)m0 * Tc + t + 1) * HP, 0, 1);
#endif
    __syncthreads();

    // --- accumulators: r, z, gi_n, gh_n (gi_n/gh_n must stay separate) ----
    v8f accR, accZ, accGN, accHN;
    if (HAS_X) {
      float br = bihP[jc] + bhhP[jc];
      float bz = bihP[512 + jc] + bhhP[512 + jc];
      float bni = bihP[1024 + jc];
      float bnh = bhhP[1024 + jc];
#pragma unroll
      for (int v = 0; v < 8; ++v) {
        accR[v] = br; accZ[v] = bz; accGN[v] = bni; accHN[v] = bnh;
      }
    } else {
      float bhr = bhhP[jc], bhz = bhhP[512 + jc], bnh = bhhP[1024 + jc];
#pragma unroll
      for (int v = 0; v < 8; ++v) {
        int m = v + half * 8;
        const float* g = gi1 + (size_t)(m0 + m) * GP;
        accR[v]  = g[jc] + bhr;
        accZ[v]  = g[512 + jc] + bhz;
        accGN[v] = g[1024 + jc];
        accHN[v] = bnh;
      }
    }

    // --- K loop: gh (+ gi for layers 2/3) via WMMA ------------------------
#pragma unroll 2
    for (int kb = 0; kb < HP; kb += 32) {
      v16bf aH = load_a_frag(Hsh, HP, kb, lane);
      v16bf bR = load_b_frag(WhhP, j0,        kb, lane);
      v16bf bZ = load_b_frag(WhhP, 512 + j0,  kb, lane);
      v16bf bN = load_b_frag(WhhP, 1024 + j0, kb, lane);
      accR  = wmma_bf16(aH, bR, accR);
      accZ  = wmma_bf16(aH, bZ, accZ);
      accHN = wmma_bf16(aH, bN, accHN);
      if (HAS_X) {
        v16bf aX = load_a_frag(Xcur, HP, kb, lane);
        v16bf cR = load_b_frag(WihP, j0,        kb, lane);
        v16bf cZ = load_b_frag(WihP, 512 + j0,  kb, lane);
        v16bf cN = load_b_frag(WihP, 1024 + j0, kb, lane);
        accR  = wmma_bf16(aX, cR, accR);
        accZ  = wmma_bf16(aX, cZ, accZ);
        accGN = wmma_bf16(aX, cN, accGN);
      }
    }

    // --- gate fusion + h update ------------------------------------------
    bf16* Hdst = Hbuf + (size_t)((t + 1) & 1) * (Bc * HP);
#pragma unroll
    for (int v = 0; v < 8; ++v) {
      int m = v + half * 8;  // C/D layout: VGPR v -> M=v (lanes<16) / v+8
      float r = sigmoidf_(accR[v]);
      float z = sigmoidf_(accZ[v]);
      float n = tanhf(accGN[v] + r * accHN[v]);
      float hp = (float)Hsh[m * HP + jc];
      float hn = (1.f - z) * n + z * hp;
      if (jc >= Hc) hn = 0.f;  // keep K-padding exactly zero
      Hdst[(size_t)(m0 + m) * HP + jc] = (bf16)hn;
      Oseq[((size_t)(m0 + m) * Tc + t) * HP + jc] =
          (bf16)(jc < Hc ? tanhf(hn) : 0.f);
    }

    // --- cross-slice step barrier (cluster split-barrier, ISA barrier -3) -
    __builtin_amdgcn_fence(__ATOMIC_RELEASE, "agent");  // flush h stores
    __syncthreads();                                    // all waves flushed
    if (threadIdx.x < 32)                               // one signal per WG
      asm volatile("s_barrier_signal -3" ::: "memory");
    asm volatile("s_barrier_wait -3" ::: "memory");
    __builtin_amdgcn_fence(__ATOMIC_ACQUIRE, "agent");  // invalidate stale h
  }
}

// ---------------------------------------------------------------------------
// Decoder: logits = relu(O3 @ Wd^T + bd), [51200,512] x [512,80], f32 out.
// One wave per 16x16 output tile.
// ---------------------------------------------------------------------------
__global__ __launch_bounds__(32) void decoder_kernel(
    const bf16* __restrict__ O3,    // [NROW, HP]
    const bf16* __restrict__ WdP,   // [VP, HP]
    const float* __restrict__ bd,   // [Vc]
    float* __restrict__ out)        // [NROW, Vc]
{
  const int lane = threadIdx.x & 31;
  const int m0 = blockIdx.x * 16;
  const int n0 = blockIdx.y * 16;
  const int nc = n0 + (lane & 15);
  const int half = lane >> 4;

  v8f acc;
  float bv = (nc < Vc) ? bd[nc] : 0.f;
#pragma unroll
  for (int v = 0; v < 8; ++v) acc[v] = bv;

#pragma unroll 4
  for (int kb = 0; kb < HP; kb += 32) {
    v16bf a = load_a_frag(O3 + (size_t)m0 * HP, HP, kb, lane);
    v16bf b = load_b_frag(WdP, n0, kb, lane);
    acc = wmma_bf16(a, b, acc);
  }

  if (nc < Vc) {
#pragma unroll
    for (int v = 0; v < 8; ++v) {
      int m = v + half * 8;
      out[(size_t)(m0 + m) * Vc + nc] = fmaxf(acc[v], 0.f);
    }
  }
}

// ---------------------------------------------------------------------------
// Host launcher
// ---------------------------------------------------------------------------
extern "C" void kernel_launch(void* const* d_in, const int* in_sizes, int n_in,
                              void* d_out, int out_size, void* d_ws, size_t ws_size,
                              hipStream_t stream) {
  const float* x     = (const float*)d_in[0];
  const float* Wih1  = (const float*)d_in[1];
  const float* Whh1  = (const float*)d_in[2];
  const float* bih1  = (const float*)d_in[3];
  const float* bhh1  = (const float*)d_in[4];
  const float* Wih2  = (const float*)d_in[5];
  const float* Whh2  = (const float*)d_in[6];
  const float* bih2  = (const float*)d_in[7];
  const float* bhh2  = (const float*)d_in[8];
  const float* Wih3  = (const float*)d_in[9];
  const float* Whh3  = (const float*)d_in[10];
  const float* bih3  = (const float*)d_in[11];
  const float* bhh3  = (const float*)d_in[12];
  const float* Wd    = (const float*)d_in[13];
  const float* bd    = (const float*)d_in[14];
  float* out = (float*)d_out;

  // ---- workspace carve-out (aligned 256B) ----
  size_t off = 0;
  auto carve = [&](size_t bytes) -> void* {
    off = (off + 255) & ~(size_t)255;
    void* p = (char*)d_ws + off;
    off += bytes;
    return p;
  };
  const size_t wsz = (size_t)GP * HP * sizeof(bf16);
  bf16* WhhP1 = (bf16*)carve(wsz);
  bf16* WhhP2 = (bf16*)carve(wsz);
  bf16* WhhP3 = (bf16*)carve(wsz);
  bf16* WihP2 = (bf16*)carve(wsz);
  bf16* WihP3 = (bf16*)carve(wsz);
  bf16* WdP   = (bf16*)carve((size_t)VP * HP * sizeof(bf16));
  float* bhhP1 = (float*)carve(GP * sizeof(float));
  float* bhhP2 = (float*)carve(GP * sizeof(float));
  float* bhhP3 = (float*)carve(GP * sizeof(float));
  float* bihP2 = (float*)carve(GP * sizeof(float));
  float* bihP3 = (float*)carve(GP * sizeof(float));
  float* gi1   = (float*)carve((size_t)Bc * GP * sizeof(float));
  bf16* Hbuf   = (bf16*)carve((size_t)2 * Bc * HP * sizeof(bf16));
  bf16* Oa     = (bf16*)carve((size_t)NROW * HP * sizeof(bf16));
  bf16* Ob     = (bf16*)carve((size_t)NROW * HP * sizeof(bf16));
  (void)ws_size; (void)in_sizes; (void)n_in; (void)out_size;

  // ---- pack weights/biases into padded bf16 WMMA layout ----
  const int TPB = 256;
  pack3_kernel<<<(GP * HP + TPB - 1) / TPB, TPB, 0, stream>>>(Whh1, WhhP1, Hc);
  pack3_kernel<<<(GP * HP + TPB - 1) / TPB, TPB, 0, stream>>>(Whh2, WhhP2, Hc);
  pack3_kernel<<<(GP * HP + TPB - 1) / TPB, TPB, 0, stream>>>(Whh3, WhhP3, Hc);
  pack3_kernel<<<(GP * HP + TPB - 1) / TPB, TPB, 0, stream>>>(Wih2, WihP2, Hc);
  pack3_kernel<<<(GP * HP + TPB - 1) / TPB, TPB, 0, stream>>>(Wih3, WihP3, Hc);
  packwd_kernel<<<(VP * HP + TPB - 1) / TPB, TPB, 0, stream>>>(Wd, WdP);
  packbias_kernel<<<(GP + TPB - 1) / TPB, TPB, 0, stream>>>(bhh1, bhhP1);
  packbias_kernel<<<(GP + TPB - 1) / TPB, TPB, 0, stream>>>(bhh2, bhhP2);
  packbias_kernel<<<(GP + TPB - 1) / TPB, TPB, 0, stream>>>(bhh3, bhhP3);
  packbias_kernel<<<(GP + TPB - 1) / TPB, TPB, 0, stream>>>(bih2, bihP2);
  packbias_kernel<<<(GP + TPB - 1) / TPB, TPB, 0, stream>>>(bih3, bihP3);
  gi1_kernel<<<(Bc * GP + TPB - 1) / TPB, TPB, 0, stream>>>(x, Wih1, bih1, gi1);
  zeroh_kernel<<<(2 * Bc * HP + TPB - 1) / TPB, TPB, 0, stream>>>(Hbuf);

  // ---- three GRU layers (h carries over: layer L+1 h0 = layer L h_last) ----
  dim3 ggrid(Bc / 16, 8);
  gru_seq_kernel<false><<<ggrid, 128, 0, stream>>>(
      gi1, Oa /*unused*/, WhhP1 /*unused*/, WhhP1, bhhP1 /*unused*/, bhhP1,
      Hbuf, Oa);
  gru_seq_kernel<true><<<ggrid, 128, 0, stream>>>(
      gi1 /*unused*/, Oa, WihP2, WhhP2, bihP2, bhhP2, Hbuf, Ob);
  gru_seq_kernel<true><<<ggrid, 128, 0, stream>>>(
      gi1 /*unused*/, Ob, WihP3, WhhP3, bihP3, bhhP3, Hbuf, Oa);

  // ---- decoder GEMM + ReLU ----
  dim3 dgrid(NROW / 16, VP / 16);
  decoder_kernel<<<dgrid, 32, 0, stream>>>(Oa, WdP, bd, out);
}
